// PointNet2InstNorm_33397665694064
// MI455X (gfx1250) — matrix operand order repaired and checked by reading, not compiled, not run on hardware
//
#include <hip/hip_runtime.h>
#include <math.h>

// ---------------------------------------------------------------------------
// PointNet++ (SSG, instance-norm MLPs) forward for MI455X (gfx1250).
// Linear layers: bf16 V_WMMA_F32_16X16X32_BF16 tiled GEMMs, double-buffered
// LDS staging. FULL template specialization for tile-divisible interior GEMMs
// (rows%32==0, cin%32==0, cout%64==0) removes all bounds/alignment branching.
// ---------------------------------------------------------------------------

#define BB 4
#define N0 8192

typedef __attribute__((ext_vector_type(16))) __bf16 bf16x16;
typedef __attribute__((ext_vector_type(8)))  __bf16 bf16x8;
typedef __attribute__((ext_vector_type(4)))  __bf16 bf16x4;
typedef __attribute__((ext_vector_type(2)))  __bf16 bf16x2;
typedef __attribute__((ext_vector_type(8)))  float  f32x8;

// ---------------- farthest point sampling (1 block per batch) --------------
__global__ __launch_bounds__(256) void fps_kernel(const float* __restrict__ pos,
                                                  int n, int m, int* __restrict__ idx_out) {
  int b = blockIdx.x;
  const float* P = pos + (size_t)b * n * 3;
  int* I = idx_out + (size_t)b * m;
  __shared__ float dsh[8192];
  __shared__ float redv[256];
  __shared__ int   redi[256];
  int t = threadIdx.x;
  float x0 = P[0], y0 = P[1], z0 = P[2];
  for (int j = t; j < n; j += 256) {
    float dx = P[3*j] - x0, dy = P[3*j+1] - y0, dz = P[3*j+2] - z0;
    dsh[j] = dx*dx + dy*dy + dz*dz;
  }
  if (t == 0) I[0] = 0;
  __syncthreads();
  for (int s = 1; s < m; s++) {
    float bv = -1.0f; int bi = 0;
    for (int j = t; j < n; j += 256) {
      float v = dsh[j];
      if (v > bv) { bv = v; bi = j; }
    }
    redv[t] = bv; redi[t] = bi;
    __syncthreads();
    for (int o = 128; o > 0; o >>= 1) {
      if (t < o && redv[t + o] > redv[t]) { redv[t] = redv[t + o]; redi[t] = redi[t + o]; }
      __syncthreads();
    }
    int nxt = redi[0];
    if (t == 0) I[s] = nxt;
    float nx = P[3*nxt], ny = P[3*nxt+1], nz = P[3*nxt+2];
    for (int j = t; j < n; j += 256) {
      float dx = P[3*j] - nx, dy = P[3*j+1] - ny, dz = P[3*j+2] - nz;
      float nd = dx*dx + dy*dy + dz*dz;
      if (nd < dsh[j]) dsh[j] = nd;
    }
    __syncthreads();
  }
}

// ---------------- gather sampled positions ---------------------------------
__global__ __launch_bounds__(256) void gather_pos_kernel(const float* __restrict__ P,
                                                         const int* __restrict__ idx,
                                                         int n, int m, float* __restrict__ out,
                                                         int total) {
  int i = blockIdx.x * 256 + threadIdx.x;
  if (i >= total) return;
  int c = i % 3; int t = (i / 3) % m; int b = i / (3 * m);
  int j = idx[b * m + t];
  out[i] = P[((size_t)b * n + j) * 3 + c];
}

// ------- ball query + gather + max aggregation (1 block per center) --------
__global__ __launch_bounds__(256) void sa_group_kernel(const float* __restrict__ x,
                                                       const float* __restrict__ pos,
                                                       const float* __restrict__ q,
                                                       int n, int m, int C, float r2,
                                                       float* __restrict__ out) {
  int ci = blockIdx.x, b = blockIdx.y;
  const float* P = pos + (size_t)b * n * 3;
  const float* X = x + (size_t)b * n * C;
  const float* Q = q + ((size_t)b * m + ci) * 3;
  float qx = Q[0], qy = Q[1], qz = Q[2];
  __shared__ unsigned flags[256];
  int t = threadIdx.x;
  flags[t] = 0u;
  __syncthreads();
  for (int j = t; j < n; j += 256) {
    float dx = P[3*j] - qx, dy = P[3*j+1] - qy, dz = P[3*j+2] - qz;
    float d2 = dx*dx + dy*dy + dz*dz;
    if (d2 <= r2) atomicOr(&flags[j >> 5], 1u << (j & 31));
  }
  __syncthreads();
  int words = (n + 31) >> 5;
  int Ct = C + 3;
  for (int c = t; c < Ct; c += 256) {
    float mx = -INFINITY;
    if (c < C) {
      for (int w = 0; w < words; w++) {
        unsigned f = flags[w];
        while (f) {
          int j = (w << 5) + __builtin_ctz(f);
          f &= f - 1;
          float v = X[(size_t)j * C + c];
          if (v > mx) mx = v;
        }
      }
    } else {
      int dcomp = c - C;
      float qd = (dcomp == 0) ? qx : (dcomp == 1 ? qy : qz);
      for (int w = 0; w < words; w++) {
        unsigned f = flags[w];
        while (f) {
          int j = (w << 5) + __builtin_ctz(f);
          f &= f - 1;
          float v = P[3*j + dcomp] - qd;
          if (v > mx) mx = v;
        }
      }
    }
    out[((size_t)b * m + ci) * Ct + c] = mx;
  }
}

// ---------------- bf16 WMMA tiled GEMM:  H = A(rows,cin) @ W(cin,cout) + b -
// block = 256 threads = 8 waves; block tile 32x64; wave tile 16x16.
// FULL=true: rows%32==0, cin%32==0, cout%64==0 -> all checks fold away.
template <bool FULL>
__global__ __launch_bounds__(256) void wmma_gemm_kernel(const float* __restrict__ A,
                                                        const float* __restrict__ W,
                                                        const float* __restrict__ bias,
                                                        float* __restrict__ Hout,
                                                        int rows, int cin, int cout, int relu) {
  int b = blockIdx.z;
  const float* Ab = A + (size_t)b * rows * cin;
  float* Ob = Hout + (size_t)b * rows * cout;
  int rowBase = blockIdx.y * 32;
  int colBase = blockIdx.x * 64;

  __shared__ __align__(16) __bf16 As[2][32][40];   // [buf][m][k]
  __shared__ __align__(32) __bf16 Bst[2][64][48];  // [buf][n][k] (W^T tile)

  int t = threadIdx.x;
  int wid = t >> 5, lane = t & 31;
  int waveM = wid & 1, waveN = wid >> 1;     // 2 x 4 wave grid
  int half = lane >> 4, lr = lane & 15;
  int ar = waveM * 16 + lr;                  // A row within block tile
  int kb = half * 8;                         // A k-base per ISA layout
  int bc = waveN * 16 + lr;                  // B col within block tile

  bool fullRowA = FULL || (rowBase + 32 <= rows);
  bool fullColB = FULL || (colBase + 64 <= cout);
  bool aligned4 = FULL || ((cin & 3) == 0);  // float4 path legal (ws is 256B-aligned)

  // per-thread fast-path coordinates
  int fr = t >> 3, fkq = (t & 7) << 2;       // A: 1 float4 per thread (32x32 tile)

  auto stageA = [&](int buf, int k0) {
    if (fullRowA && (FULL || (k0 + 32 <= cin))) {
      const float* src = &Ab[(size_t)(rowBase + fr) * cin + k0 + fkq];
      if (aligned4) {
        float4 v = *(const float4*)src;
        bf16x4 pv = {(__bf16)v.x, (__bf16)v.y, (__bf16)v.z, (__bf16)v.w};
        *(bf16x4*)&As[buf][fr][fkq] = pv;
      } else {
        bf16x4 pv = {(__bf16)src[0], (__bf16)src[1], (__bf16)src[2], (__bf16)src[3]};
        *(bf16x4*)&As[buf][fr][fkq] = pv;
      }
    } else {
      for (int i = t; i < 32 * 32; i += 256) {
        int r = i >> 5, k = i & 31;
        int gr = rowBase + r, gk = k0 + k;
        float v = (gr < rows && gk < cin) ? Ab[(size_t)gr * cin + gk] : 0.f;
        As[buf][r][k] = (__bf16)v;
      }
    }
  };
  auto stageB = [&](int buf, int k0) {
    if (fullColB && (FULL || (k0 + 32 <= cin))) {
#pragma unroll
      for (int pi = 0; pi < 4; pi++) {
        int i = t + (pi << 8);
        int n = i & 63, kk = (i >> 6) << 1;
        const float* src = &W[(size_t)(k0 + kk) * cout + colBase + n];
        bf16x2 pv = {(__bf16)src[0], (__bf16)src[cout]};
        *(bf16x2*)&Bst[buf][n][kk] = pv;
      }
    } else {
      for (int i = t; i < 32 * 64; i += 256) {
        int k = i >> 6, n = i & 63;
        int gk = k0 + k, gn = colBase + n;
        float v = (gk < cin && gn < cout) ? W[(size_t)gk * cout + gn] : 0.f;
        Bst[buf][n][k] = (__bf16)v;
      }
    }
  };

  f32x8 acc = {0.f, 0.f, 0.f, 0.f, 0.f, 0.f, 0.f, 0.f};

  int nk = (cin + 31) >> 5;
  stageA(0, 0);
  stageB(0, 0);
  __syncthreads();
  for (int it = 0; it < nk; ++it) {
    int cur = it & 1;
    int k0n = (it + 1) << 5;
    if (k0n < cin) {
      // prefetch tile-after-next of the A stream into near caches
      if (fullRowA && (k0n + 32 + fkq) < cin)
        __builtin_prefetch(&Ab[(size_t)(rowBase + fr) * cin + k0n + 32 + fkq], 0, 3);
      stageA(cur ^ 1, k0n);
      stageB(cur ^ 1, k0n);
    }
    // A fragment per ISA 16-bit A 16x32 layout (lane half -> K+8; elems 0..7 ->
    // K kb..kb+7, elems 8..15 -> K kb+16..kb+23): two 16B LDS loads.
    bf16x8 a0 = *(const bf16x8*)&As[cur][ar][kb];
    bf16x8 a1 = *(const bf16x8*)&As[cur][ar][kb + 16];
    bf16x16 af = __builtin_shufflevector(a0, a1, 0,1,2,3,4,5,6,7,8,9,10,11,12,13,14,15);
    // B fragment: same K layout as A (kb..kb+7, kb+16..kb+23): two 16B LDS loads
    bf16x8 b0 = *(const bf16x8*)&Bst[cur][bc][kb];
    bf16x8 b1 = *(const bf16x8*)&Bst[cur][bc][kb + 16];
    bf16x16 bfv = __builtin_shufflevector(b0, b1, 0,1,2,3,4,5,6,7,8,9,10,11,12,13,14,15);
    acc = __builtin_amdgcn_wmma_f32_16x16x32_bf16(false, af, false, bfv,
                                                  (short)0, acc, false, false);
    __syncthreads();
  }

  // C/D layout: lanes 0-15 N=lane M=v; lanes 16-31 N=lane-16 M=v+8
  int col = colBase + waveN * 16 + lr;
  if (FULL || col < cout) {
    float bb = bias ? bias[col] : 0.f;
#pragma unroll
    for (int v = 0; v < 8; v++) {
      int row = rowBase + waveM * 16 + v + half * 8;
      if (FULL || row < rows) {
        float o = acc[v] + bb;
        if (relu) o = fmaxf(o, 0.f);
        Ob[(size_t)row * cout + col] = o;
      }
    }
  }
}

// ---------------- instance norm: per (batch, channel) over rows ------------
__global__ __launch_bounds__(256) void in_stats_kernel(const float* __restrict__ H,
                                                       int rows, int cols,
                                                       float* __restrict__ mean,
                                                       float* __restrict__ rstd) {
  int c = blockIdx.x, b = blockIdx.y;
  const float* p = H + (size_t)b * rows * cols + c;
  float s = 0.f, s2 = 0.f;
  for (int r = threadIdx.x; r < rows; r += 256) {
    float v = p[(size_t)r * cols];
    s += v; s2 += v * v;
  }
  __shared__ float sh[256], sq[256];
  int t = threadIdx.x;
  sh[t] = s; sq[t] = s2;
  __syncthreads();
  for (int o = 128; o > 0; o >>= 1) {
    if (t < o) { sh[t] += sh[t + o]; sq[t] += sq[t + o]; }
    __syncthreads();
  }
  if (t == 0) {
    float inv = 1.f / (float)rows;
    float mu = sh[0] * inv;
    float var = sq[0] * inv - mu * mu;
    mean[b * cols + c] = mu;
    rstd[b * cols + c] = rsqrtf(fmaxf(var, 0.f) + 1e-5f);
  }
}

__global__ __launch_bounds__(256) void in_apply_kernel(float* __restrict__ H,
                                                       const float* __restrict__ mean,
                                                       const float* __restrict__ rstd,
                                                       int rows, int cols, int total) {
  int i = blockIdx.x * 256 + threadIdx.x;
  if (i >= total) return;
  int c = i % cols;
  int b = i / (rows * cols);
  float v = (H[i] - mean[b * cols + c]) * rstd[b * cols + c];
  H[i] = v > 0.f ? v : 0.f;
}

// ---------------- kNN (k<=3) indices + inverse-d2 weights ------------------
__global__ __launch_bounds__(256) void knn_kernel(const float* __restrict__ pt,
                                                  const float* __restrict__ ps,
                                                  int nt, int ns, int k,
                                                  int* __restrict__ idx, float* __restrict__ w) {
  int i = blockIdx.x * 256 + threadIdx.x;
  int b = blockIdx.y;
  if (i >= nt) return;
  const float* T = pt + ((size_t)b * nt + i) * 3;
  const float* S = ps + (size_t)b * ns * 3;
  float tx = T[0], ty = T[1], tz = T[2];
  float d0 = 3.4e38f, d1 = 3.4e38f, d2m = 3.4e38f;
  int i0 = 0, i1 = 0, i2 = 0;
  for (int j = 0; j < ns; j++) {
    float dx = S[3*j] - tx, dy = S[3*j+1] - ty, dz = S[3*j+2] - tz;
    float dd = dx*dx + dy*dy + dz*dz;
    if (dd < d0)      { d2m = d1; i2 = i1; d1 = d0; i1 = i0; d0 = dd; i0 = j; }
    else if (dd < d1) { d2m = d1; i2 = i1; d1 = dd; i1 = j; }
    else if (dd < d2m){ d2m = dd; i2 = j; }
  }
  size_t base = ((size_t)b * nt + i) * 3;
  int   bi[3] = {i0, i1, i2};
  float bd[3] = {d0, d1, d2m};
  for (int kk = 0; kk < k; kk++) {
    idx[base + kk] = bi[kk];
    w[base + kk] = 1.f / fmaxf(bd[kk], 1e-16f);
  }
}

// interpolated features into left columns of concat buffer (stride outStride)
__global__ __launch_bounds__(256) void knn_gather_kernel(const float* __restrict__ Xs,
                                                         const int* __restrict__ idx,
                                                         const float* __restrict__ w,
                                                         int nt, int ns, int k, int C,
                                                         int outStride, float* __restrict__ out,
                                                         int total) {
  int i = blockIdx.x * 256 + threadIdx.x;
  if (i >= total) return;
  int c = i % C; int t = (i / C) % nt; int b = i / (C * nt);
  const int*   I  = idx + ((size_t)b * nt + t) * 3;
  const float* Wt = w   + ((size_t)b * nt + t) * 3;
  float num = 0.f, den = 0.f;
  for (int kk = 0; kk < k; kk++) {
    float ww = Wt[kk];
    num += ww * Xs[((size_t)b * ns + I[kk]) * C + c];
    den += ww;
  }
  out[((size_t)b * nt + t) * outStride + c] = num / den;
}

// copy skip features into right columns of concat buffer
__global__ __launch_bounds__(256) void copy_cols_kernel(const float* __restrict__ src,
                                                        int nt, int Cs, int dstStride, int dstOff,
                                                        float* __restrict__ dst, int total) {
  int i = blockIdx.x * 256 + threadIdx.x;
  if (i >= total) return;
  int c = i % Cs; int t = (i / Cs) % nt; int b = i / (Cs * nt);
  dst[((size_t)b * nt + t) * dstStride + dstOff + c] = src[i];
}

// ---------------------------------------------------------------------------
extern "C" void kernel_launch(void* const* d_in, const int* in_sizes, int n_in,
                              void* d_out, int out_size, void* d_ws, size_t ws_size,
                              hipStream_t stream) {
  (void)n_in; (void)out_size; (void)ws_size;

  static const int M[6]  = {8192, 1638, 491, 147, 44, 13};
  static const int CH[6] = {3, 128, 128, 256, 512, 1024};
  static const float R2[5] = {0.0025f, 0.01f, 0.04f, 0.16f, 0.64f};
  static const int saCh[5][4] = {{6,64,64,128},{131,128,128,128},{259,256,256,256},
                                 {515,512,512,512},{515,512,512,1024}};
  // fp index 0 = fp5 ... 4 = fp1
  static const int fpCh[5][4] = {{1536,512,512,0},{768,256,256,0},{384,256,256,0},
                                 {384,256,128,0},{131,128,128,128}};
  static const int fpK[5] = {1, 3, 3, 3, 3};
  static const int fpL[5] = {2, 2, 2, 2, 3};

  auto gp = [&](int i) { return (const float*)d_in[i]; };
  const float *x_in = nullptr, *pos_in = nullptr;
  const float *saW[5][3], *saB[5][3];
  const float *fpW[5][3], *fpB[5][3];
  const float *linW[3], *linB[3];

  // Runtime detection of flatten order: insertion order puts x (4*8192*3=98304)
  // first; jax tree-flatten sorted order puts fp1 W0 (131*128=16768) first.
  bool xFirst = (in_sizes[0] == BB * N0 * 3);
  int p = 0;
  if (xFirst) {
    x_in = gp(p++); pos_in = gp(p++);
    for (int i = 0; i < 5; i++)
      for (int l = 0; l < 3; l++) { saW[i][l] = gp(p++); saB[i][l] = gp(p++); }
    for (int i = 0; i < 5; i++)                           // fp5 .. fp1
      for (int l = 0; l < fpL[i]; l++) { fpW[i][l] = gp(p++); fpB[i][l] = gp(p++); }
    for (int i = 0; i < 3; i++) { linW[i] = gp(p++); linB[i] = gp(p++); }
  } else {
    for (int i = 4; i >= 0; i--)                          // fp1 .. fp5 (sorted)
      for (int l = 0; l < fpL[i]; l++) { fpW[i][l] = gp(p++); fpB[i][l] = gp(p++); }
    for (int i = 0; i < 3; i++) { linW[i] = gp(p++); linB[i] = gp(p++); }
    for (int i = 0; i < 5; i++)
      for (int l = 0; l < 3; l++) { saW[i][l] = gp(p++); saB[i][l] = gp(p++); }
    pos_in = gp(p++); x_in = gp(p++);
  }

  // ---- workspace bump allocator ----
  char* wp = (char*)d_ws;
  size_t off = 0;
  auto walloc = [&](size_t bytes) -> void* {
    void* r = wp + off;
    off = (off + bytes + 255) & ~(size_t)255;
    return r;
  };

  int* idxL[5]; float* posL[6]; float* gL[5]; float* hA[6];
  posL[0] = nullptr; hA[0] = nullptr;
  for (int i = 0; i < 5; i++) {
    idxL[i]     = (int*)  walloc((size_t)BB * M[i+1] * sizeof(int));
    posL[i + 1] = (float*)walloc((size_t)BB * M[i+1] * 3 * sizeof(float));
    gL[i]       = (float*)walloc((size_t)BB * M[i+1] * (CH[i] + 3) * sizeof(float));
    hA[i + 1]   = (float*)walloc((size_t)BB * M[i+1] * CH[i+1] * sizeof(float));
  }
  size_t BUFE = (size_t)BB * 8192 * 132;   // covers max rows*cols (8192 x 131)
  float* bufs[2];
  bufs[0] = (float*)walloc(BUFE * sizeof(float));
  bufs[1] = (float*)walloc(BUFE * sizeof(float));
  float* meanB = (float*)walloc((size_t)BB * 1024 * sizeof(float));
  float* rstdB = (float*)walloc((size_t)BB * 1024 * sizeof(float));
  int*   knnI  = (int*)  walloc((size_t)BB * N0 * 3 * sizeof(int));
  float* knnWt = (float*)walloc((size_t)BB * N0 * 3 * sizeof(float));

  auto gemm = [&](const float* A, const float* W, const float* bias, float* H,
                  int rows, int cin, int cout, int relu) {
    dim3 g((cout + 63) / 64, (rows + 31) / 32, BB);
    bool full = ((rows & 31) == 0) && ((cin & 31) == 0) && ((cout & 63) == 0);
    if (full)
      wmma_gemm_kernel<true><<<g, 256, 0, stream>>>(A, W, bias, H, rows, cin, cout, relu);
    else
      wmma_gemm_kernel<false><<<g, 256, 0, stream>>>(A, W, bias, H, rows, cin, cout, relu);
  };
  auto inorm = [&](float* H, int rows, int cols) {
    dim3 g(cols, BB);
    in_stats_kernel<<<g, 256, 0, stream>>>(H, rows, cols, meanB, rstdB);
    int total = BB * rows * cols;
    in_apply_kernel<<<(total + 255) / 256, 256, 0, stream>>>(H, meanB, rstdB, rows, cols, total);
  };

  // ================= SA (set abstraction) levels =================
  const float* curX = x_in;
  const float* curP = pos_in;
  for (int i = 0; i < 5; i++) {
    int n = M[i], m = M[i + 1], C = CH[i];
    fps_kernel<<<BB, 256, 0, stream>>>(curP, n, m, idxL[i]);
    int tp = BB * m * 3;
    gather_pos_kernel<<<(tp + 255) / 256, 256, 0, stream>>>(curP, idxL[i], n, m, posL[i + 1], tp);
    dim3 gg(m, BB);
    sa_group_kernel<<<gg, 256, 0, stream>>>(curX, curP, posL[i + 1], n, m, C, R2[i], gL[i]);
    const float* src = gL[i];
    int cin = C + 3;
    float* outs[3] = {bufs[0], bufs[1], hA[i + 1]};
    for (int l = 0; l < 3; l++) {
      int cout = saCh[i][l + 1];
      gemm(src, saW[i][l], saB[i][l], outs[l], m, cin, cout, 0);
      inorm(outs[l], m, cout);
      src = outs[l]; cin = cout;
    }
    curX = hA[i + 1];
    curP = posL[i + 1];
  }

  // ================= FP (feature propagation) levels =================
  const float* cx = hA[5];
  const float* cp = posL[5];
  int cc = 1024, cn = M[5];
  int curBuf = -1;  // cx not in ping-pong buffers yet
  for (int j = 0; j < 5; j++) {          // j: 0=fp5 .. 4=fp1
    int lvl = 4 - j;
    int nt = M[lvl];
    const float* tpz  = (lvl == 0) ? pos_in : posL[lvl];
    const float* skip = (lvl == 0) ? x_in   : hA[lvl];
    int cs = CH[lvl];
    int Ctot = cc + cs;                  // == fpCh[j][0]
    int k = fpK[j];

    dim3 kg((nt + 255) / 256, BB);
    knn_kernel<<<kg, 256, 0, stream>>>(tpz, cp, nt, cn, k, knnI, knnWt);

    int catB = (curBuf == 0) ? 1 : 0;
    float* cat = bufs[catB];
    int t1 = BB * nt * cc;
    knn_gather_kernel<<<(t1 + 255) / 256, 256, 0, stream>>>(cx, knnI, knnWt, nt, cn, k, cc,
                                                            Ctot, cat, t1);
    int t2 = BB * nt * cs;
    copy_cols_kernel<<<(t2 + 255) / 256, 256, 0, stream>>>(skip, nt, cs, Ctot, cc, cat, t2);

    const float* srcp = cat;
    int cin = Ctot, s = catB;
    for (int l = 0; l < fpL[j]; l++) {
      int cout = fpCh[j][l + 1];
      int d = 1 - s;
      gemm(srcp, fpW[j][l], fpB[j][l], bufs[d], nt, cin, cout, 0);
      inorm(bufs[d], nt, cout);
      srcp = bufs[d]; cin = cout; s = d;
    }
    cx = bufs[s]; curBuf = s; cc = cin; cn = nt; cp = tpz;
  }

  // ================= head: lin1(relu) -> lin2 -> lin3 =================
  float* outF = (float*)d_out;
  int d1 = 1 - curBuf;
  gemm(cx,        linW[0], linB[0], bufs[d1],     N0, 128, 128, 1);
  gemm(bufs[d1],  linW[1], linB[1], bufs[curBuf], N0, 128, 128, 0);
  gemm(bufs[curBuf], linW[2], linB[2], outF,      N0, 128, 128, 0);
}
